// MultiHeadAttention_64132451664595
// MI455X (gfx1250) — matrix operand order repaired and checked by reading, not compiled, not run on hardware
//
#include <hip/hip_runtime.h>

typedef __attribute__((ext_vector_type(16))) __bf16 v16bf;
typedef __attribute__((ext_vector_type(8)))  float  v8f;
typedef __attribute__((ext_vector_type(4)))  int    i32x4;

union FragB {
  v16bf v;
  uint4 q[2];
  unsigned short u[16];
};

union Pack8 {
  unsigned short u[8];
  uint4 q;
};

__device__ __forceinline__ unsigned short f32_to_bf16(float f) {
  unsigned int u = __float_as_uint(f);
  u += 0x7FFFu + ((u >> 16) & 1u);   // round-to-nearest-even
  return (unsigned short)(u >> 16);
}

__device__ __forceinline__ v8f wmma_bf16(const FragB& a, const FragB& b, v8f c) {
  return __builtin_amdgcn_wmma_f32_16x16x32_bf16(
      false, a.v, false, b.v, (short)0, c, false, false);
}

// ---- async 16-byte global->LDS copy (ASYNCcnt path on gfx1250) -------------
#if defined(__has_builtin)
#if __has_builtin(__builtin_amdgcn_global_load_async_to_lds_b128)
#define ASYNC_MODE 1
#elif __has_builtin(__builtin_amdgcn_load_to_lds)
#define ASYNC_MODE 2
#else
#define ASYNC_MODE 0
#endif
#else
#define ASYNC_MODE 0
#endif

__device__ __forceinline__ void async_copy16(const unsigned short* g, unsigned short* l) {
#if ASYNC_MODE == 1
  __builtin_amdgcn_global_load_async_to_lds_b128(
      (__attribute__((address_space(1))) i32x4*)g,
      (__attribute__((address_space(3))) i32x4*)l, 0, 0);
#elif ASYNC_MODE == 2
  __builtin_amdgcn_load_to_lds((void*)g,
      (__attribute__((address_space(3))) void*)l, 16, 0, 0);
#else
  *(uint4*)l = *(const uint4*)g;
#endif
}

__device__ __forceinline__ void wait_async() {
#if ASYNC_MODE == 1
  asm volatile("s_wait_asynccnt 0" ::: "memory");
#elif ASYNC_MODE == 2
  asm volatile("s_wait_asynccnt 0\ns_wait_loadcnt 0x0\ns_wait_dscnt 0x0" ::: "memory");
#endif
}

// ---------------------------------------------------------------------------
// One-time f32 -> bf16 conversion (same layout). 8 elements / thread.
// ---------------------------------------------------------------------------
__global__ __launch_bounds__(256)
void cvt_f32_bf16(const float* __restrict__ src, unsigned short* __restrict__ dst) {
  const size_t i = ((size_t)blockIdx.x * 256 + threadIdx.x) * 8;
  float4 f0 = *(const float4*)(src + i);
  float4 f1 = *(const float4*)(src + i + 4);
  Pack8 p;
  p.u[0] = f32_to_bf16(f0.x); p.u[1] = f32_to_bf16(f0.y);
  p.u[2] = f32_to_bf16(f0.z); p.u[3] = f32_to_bf16(f0.w);
  p.u[4] = f32_to_bf16(f1.x); p.u[5] = f32_to_bf16(f1.y);
  p.u[6] = f32_to_bf16(f1.z); p.u[7] = f32_to_bf16(f1.w);
  *(uint4*)&dst[i] = p.q;
}

// ---------------------------------------------------------------------------
// One-time weight transpose+convert: W f32 [1024 k][1024 n] -> Wt bf16 [n][k].
// 64x64 tiles through LDS; both global phases coalesced.
// ---------------------------------------------------------------------------
__global__ __launch_bounds__(256)
void transpose_w_bf16(const float* __restrict__ W, unsigned short* __restrict__ Wt) {
  __shared__ __align__(16) unsigned short Ls[64 * 80]; // [n][k], padded stride
  const int k0 = blockIdx.x * 64;
  const int n0 = blockIdx.y * 64;
  const int t  = threadIdx.x;
  #pragma unroll
  for (int j = 0; j < 4; ++j) {
    const int c  = t + j * 256;      // 1024 float4-chunks over [64][16]
    const int kr = c >> 4;
    const int nc = (c & 15) * 4;
    float4 f = *(const float4*)&W[(size_t)(k0 + kr) * 1024 + n0 + nc];
    Ls[(nc + 0) * 80 + kr] = f32_to_bf16(f.x);
    Ls[(nc + 1) * 80 + kr] = f32_to_bf16(f.y);
    Ls[(nc + 2) * 80 + kr] = f32_to_bf16(f.z);
    Ls[(nc + 3) * 80 + kr] = f32_to_bf16(f.w);
  }
  __syncthreads();
  #pragma unroll
  for (int j = 0; j < 2; ++j) {
    const int c  = t + j * 256;      // 512 uint4-chunks over [64][8]
    const int nr = c >> 3;
    const int kc = (c & 7) * 8;
    *(uint4*)&Wt[(size_t)(n0 + nr) * 1024 + k0 + kc] = *(const uint4*)&Ls[nr * 80 + kc];
  }
}

// ---------------------------------------------------------------------------
// GEMM: Out = Xb[MxK] * Wt^T + bias    (Xb bf16 row-major, Wt bf16 [n][k])
//   mode 0 : Out f32 row-major [MxN]
//   mode 1 : Out bf16 per-head [B*H][S][64]
//   mode 2 : Out bf16 per-head [B*H][64][S] (transposed, for V)
// Block 256 thr = 8 waves; block tile 128x64; wave tile 32x32; K-chunk 32.
// Double-buffered async global->LDS staging.
// ---------------------------------------------------------------------------
__global__ __launch_bounds__(256)
void gemm_bf16(const unsigned short* __restrict__ Xb,
               const unsigned short* __restrict__ Wt,
               const float* __restrict__ bias,
               void* __restrict__ Out, const int mode,
               const int M, const int N, const int K)
{
  __shared__ __align__(16) unsigned short As[2][128 * 32]; // [m][k]
  __shared__ __align__(16) unsigned short Bs[2][64 * 32];  // [n][k]
  const int t    = threadIdx.x;
  const int lane = t & 31;
  const int w    = t >> 5;
  const int m0   = blockIdx.x * 128;
  const int n0   = blockIdx.y * 64;
  const int wm   = (w & 3) * 32;
  const int wn   = (w >> 2) * 32;
  const int m    = lane & 15;
  const int hi   = lane >> 4;
  const int kb   = hi * 8;
  const int kb2  = hi * 16;

  // per-thread staging coords (16B chunks)
  const int ar0 = (2 * t + 0) >> 2, ac0 = ((2 * t + 0) & 3) * 8; // A: [128][4]
  const int ar1 = (2 * t + 1) >> 2, ac1 = ((2 * t + 1) & 3) * 8;
  const int br  = t >> 2,           bc  = (t & 3) * 8;           // B: [64][4]
  const unsigned short* Ag0 = &Xb[(size_t)(m0 + ar0) * K + ac0];
  const unsigned short* Ag1 = &Xb[(size_t)(m0 + ar1) * K + ac1];
  const unsigned short* Bg  = &Wt[(size_t)(n0 + br) * K + bc];

  async_copy16(Ag0, &As[0][ar0 * 32 + ac0]);
  async_copy16(Ag1, &As[0][ar1 * 32 + ac1]);
  async_copy16(Bg,  &Bs[0][br * 32 + bc]);

  v8f c00 = {}, c01 = {}, c10 = {}, c11 = {};
  const int niter = K / 32;
  for (int it = 0; it < niter; ++it) {
    const int cur = it & 1;
    wait_async();
    __syncthreads();
    if (it + 1 < niter) {
      const int kc = (it + 1) * 32;
      async_copy16(Ag0 + kc, &As[cur ^ 1][ar0 * 32 + ac0]);
      async_copy16(Ag1 + kc, &As[cur ^ 1][ar1 * 32 + ac1]);
      async_copy16(Bg + kc,  &Bs[cur ^ 1][br * 32 + bc]);
    }
    FragB a0, a1, b0, b1;
    a0.q[0] = *(const uint4*)&As[cur][(wm + m) * 32 + kb];
    a0.q[1] = *(const uint4*)&As[cur][(wm + m) * 32 + kb + 16];
    a1.q[0] = *(const uint4*)&As[cur][(wm + 16 + m) * 32 + kb];
    a1.q[1] = *(const uint4*)&As[cur][(wm + 16 + m) * 32 + kb + 16];
    b0.q[0] = *(const uint4*)&Bs[cur][(wn + m) * 32 + kb];
    b0.q[1] = *(const uint4*)&Bs[cur][(wn + m) * 32 + kb + 16];
    b1.q[0] = *(const uint4*)&Bs[cur][(wn + 16 + m) * 32 + kb];
    b1.q[1] = *(const uint4*)&Bs[cur][(wn + 16 + m) * 32 + kb + 16];
    c00 = wmma_bf16(a0, b0, c00);
    c01 = wmma_bf16(a0, b1, c01);
    c10 = wmma_bf16(a1, b0, c10);
    c11 = wmma_bf16(a1, b1, c11);
    __syncthreads();
  }

  // Epilogue. C element i -> row = i + hi*8, col = lane&15.
  #pragma unroll
  for (int mt = 0; mt < 2; ++mt) {
    v8f* cl = mt ? &c10 : &c00;
    v8f* cr = mt ? &c11 : &c01;
    #pragma unroll
    for (int i = 0; i < 8; ++i) {
      const int gm  = m0 + wm + mt * 16 + hi * 8 + i;
      const int gn0 = n0 + wn + m;
      const int gn1 = gn0 + 16;
      const float v0f = (*cl)[i] + bias[gn0];
      const float v1f = (*cr)[i] + bias[gn1];
      if (mode == 0) {
        float* Of = (float*)Out;
        Of[(size_t)gm * N + gn0] = v0f;
        Of[(size_t)gm * N + gn1] = v1f;
      } else {
        unsigned short* Ob = (unsigned short*)Out;
        const size_t bh0 = ((size_t)(gm >> 11) * 16 + (gn0 >> 6)) << 17;
        const size_t bh1 = ((size_t)(gm >> 11) * 16 + (gn1 >> 6)) << 17;
        if (mode == 1) {
          Ob[bh0 + ((size_t)(gm & 2047) << 6) + (gn0 & 63)] = f32_to_bf16(v0f);
          Ob[bh1 + ((size_t)(gm & 2047) << 6) + (gn1 & 63)] = f32_to_bf16(v1f);
        } else {
          Ob[bh0 + ((size_t)(gn0 & 63) << 11) + (gm & 2047)] = f32_to_bf16(v0f);
          Ob[bh1 + ((size_t)(gn1 & 63) << 11) + (gm & 2047)] = f32_to_bf16(v1f);
        }
      }
    }
  }
}

// ---------------------------------------------------------------------------
// Flash attention. Block = 8 waves, all on the SAME head; wave owns 16 q-rows.
// K/V tiles (32 keys) staged in LDS via async copy, double-buffered, shared
// by all 8 waves. Q: [B*H][2048][64] bf16.  Vt: [B*H][64][2048] bf16.
// ---------------------------------------------------------------------------
__global__ __launch_bounds__(256)
void flash_attn(const unsigned short* __restrict__ Q,
                const unsigned short* __restrict__ Kc,
                const unsigned short* __restrict__ Vt,
                unsigned short* __restrict__ Ctx)
{
  __shared__ __align__(16) unsigned short Ks[2][32 * 64]; // [j][d]
  __shared__ __align__(16) unsigned short Vs[2][64 * 32]; // [d][j]
  __shared__ __align__(16) unsigned short Ps[8][16 * 32]; // per-wave P staging
  const int t    = threadIdx.x;
  const int lane = t & 31;
  const int w    = t >> 5;
  const int bh   = blockIdx.x >> 4;                 // 64 heads, 16 blocks each
  const int qb   = (blockIdx.x & 15) * 128 + w * 16;
  const unsigned short* Qh = Q  + ((size_t)bh << 17);
  const unsigned short* Kh = Kc + ((size_t)bh << 17);
  const unsigned short* Vh = Vt + ((size_t)bh << 17);
  const int m   = lane & 15;
  const int hi  = lane >> 4;
  const int kb  = hi * 8;
  const int kb2 = hi * 16;

  // staging coords: K tile 32x64 (row 128B), V tile 64x32 (row 64B)
  const int krow = t >> 3, kcol = (t & 7) * 8;
  const int vrow = t >> 2, vcol = (t & 3) * 8;
  const unsigned short* Kg = &Kh[(size_t)krow * 64 + kcol];
  const unsigned short* Vg = &Vh[(size_t)vrow * 2048 + vcol];

  // Preload Q A-fragments (d-chunks 0..31, 32..63)
  FragB qa0, qa1;
  {
    const unsigned short* qr = &Qh[(qb + m) * 64];
    qa0.q[0] = *(const uint4*)(qr + kb);
    qa0.q[1] = *(const uint4*)(qr + kb + 16);
    qa1.q[0] = *(const uint4*)(qr + 32 + kb);
    qa1.q[1] = *(const uint4*)(qr + 32 + kb + 16);
  }

  async_copy16(Kg, &Ks[0][krow * 64 + kcol]);
  async_copy16(Vg, &Vs[0][vrow * 32 + vcol]);

  float mrun[8], lrun[8];
  v8f o0 = {}, o1 = {}, o2 = {}, o3 = {};
  #pragma unroll
  for (int i = 0; i < 8; ++i) { mrun[i] = -3.0e38f; lrun[i] = 0.0f; }

  for (int it = 0; it < 64; ++it) {            // 2048 keys / 32
    const int cur = it & 1;
    wait_async();
    __syncthreads();
    if (it + 1 < 64) {
      const int j1 = (it + 1) * 32;
      async_copy16(Kg + (size_t)j1 * 64, &Ks[cur ^ 1][krow * 64 + kcol]);
      async_copy16(Vg + j1,              &Vs[cur ^ 1][vrow * 32 + vcol]);
    }

    FragB k0a, k0b, k1a, k1b;
    const unsigned short* kr0 = &Ks[cur][m * 64];
    const unsigned short* kr1 = &Ks[cur][(16 + m) * 64];
    k0a.q[0] = *(const uint4*)(kr0 + kb);      k0a.q[1] = *(const uint4*)(kr0 + kb + 16);
    k0b.q[0] = *(const uint4*)(kr0 + 32 + kb); k0b.q[1] = *(const uint4*)(kr0 + 32 + kb + 16);
    k1a.q[0] = *(const uint4*)(kr1 + kb);      k1a.q[1] = *(const uint4*)(kr1 + kb + 16);
    k1b.q[0] = *(const uint4*)(kr1 + 32 + kb); k1b.q[1] = *(const uint4*)(kr1 + 32 + kb + 16);

    v8f s0 = {}, s1 = {};
    s0 = wmma_bf16(qa0, k0a, s0);
    s0 = wmma_bf16(qa1, k0b, s0);
    s1 = wmma_bf16(qa0, k1a, s1);
    s1 = wmma_bf16(qa1, k1b, s1);

    float rmax[8];
    #pragma unroll
    for (int i = 0; i < 8; ++i) {
      s0[i] *= 0.125f;  // 1/sqrt(64)
      s1[i] *= 0.125f;
      rmax[i] = fmaxf(s0[i], s1[i]);
    }
    #pragma unroll
    for (int off = 1; off < 16; off <<= 1) {
      #pragma unroll
      for (int i = 0; i < 8; ++i)
        rmax[i] = fmaxf(rmax[i], __shfl_xor(rmax[i], off, 32));
    }
    float alpha[8], rs[8], p0[8], p1[8];
    #pragma unroll
    for (int i = 0; i < 8; ++i) {
      const float mnew = fmaxf(mrun[i], rmax[i]);
      alpha[i] = __expf(mrun[i] - mnew);
      mrun[i]  = mnew;
      p0[i] = __expf(s0[i] - mnew);
      p1[i] = __expf(s1[i] - mnew);
      rs[i] = p0[i] + p1[i];
    }
    #pragma unroll
    for (int off = 1; off < 16; off <<= 1) {
      #pragma unroll
      for (int i = 0; i < 8; ++i)
        rs[i] += __shfl_xor(rs[i], off, 32);
    }
    #pragma unroll
    for (int i = 0; i < 8; ++i) {
      lrun[i] = lrun[i] * alpha[i] + rs[i];
      o0[i] *= alpha[i]; o1[i] *= alpha[i];
      o2[i] *= alpha[i]; o3[i] *= alpha[i];
    }

    // Re-shape P (C layout) -> A-fragment layout via wave-private LDS patch
    unsigned short* ps = Ps[w];
    #pragma unroll
    for (int i = 0; i < 8; ++i) {
      ps[(hi * 8 + i) * 32 + m]      = f32_to_bf16(p0[i]);
      ps[(hi * 8 + i) * 32 + 16 + m] = f32_to_bf16(p1[i]);
    }
    asm volatile("s_wait_dscnt 0" ::: "memory");
    FragB pa;
    pa.q[0] = *(const uint4*)&ps[m * 32 + kb];
    pa.q[1] = *(const uint4*)&ps[m * 32 + kb + 16];

    #pragma unroll
    for (int tt = 0; tt < 4; ++tt) {
      FragB vb;
      const unsigned short* vr = &Vs[cur][(tt * 16 + m) * 32 + kb];
      vb.q[0] = *(const uint4*)(vr);
      vb.q[1] = *(const uint4*)(vr + 16);
      v8f& ot = (tt == 0) ? o0 : (tt == 1) ? o1 : (tt == 2) ? o2 : o3;
      ot = wmma_bf16(pa, vb, ot);
    }
    __syncthreads();
  }

  // Epilogue: ctx = O / l, to [B][S][H*64] bf16 for the output GEMM
  const int b = bh >> 4;
  const int h = bh & 15;
  #pragma unroll
  for (int i = 0; i < 8; ++i) {
    const int   row  = qb + hi * 8 + i;
    const float inv  = 1.0f / lrun[i];
    const size_t base = ((size_t)b * 2048 + row) * 1024 + h * 64;
    Ctx[base + 0 * 16 + m] = f32_to_bf16(o0[i] * inv);
    Ctx[base + 1 * 16 + m] = f32_to_bf16(o1[i] * inv);
    Ctx[base + 2 * 16 + m] = f32_to_bf16(o2[i] * inv);
    Ctx[base + 3 * 16 + m] = f32_to_bf16(o3[i] * inv);
  }
}

extern "C" void kernel_launch(void* const* d_in, const int* in_sizes, int n_in,
                              void* d_out, int out_size, void* d_ws, size_t ws_size,
                              hipStream_t stream) {
  (void)in_sizes; (void)n_in; (void)out_size; (void)ws_size;
  const float* query = (const float*)d_in[0];
  const float* key   = (const float*)d_in[1];
  const float* value = (const float*)d_in[2];
  const float* Wq    = (const float*)d_in[3];
  const float* bq    = (const float*)d_in[4];
  const float* Wk    = (const float*)d_in[5];
  const float* bk    = (const float*)d_in[6];
  const float* Wv    = (const float*)d_in[7];
  const float* bv    = (const float*)d_in[8];
  const float* Wo    = (const float*)d_in[9];
  const float* bo    = (const float*)d_in[10];

  const int M = 4 * 2048, N = 1024, K = 1024;
  const size_t MK = (size_t)M * K;      // 8Mi elements
  const size_t KK = (size_t)K * N;      // 1Mi elements per weight

  unsigned short* xb  = (unsigned short*)d_ws;   // 16 MB, reused; aliases ctx
  unsigned short* wt  = xb + MK;                 // 4 x 2 MB transposed weights
  unsigned short* qws = wt + 4 * KK;             // [B*H][S][64]
  unsigned short* kws = qws + MK;                // [B*H][S][64]
  unsigned short* vws = kws + MK;                // [B*H][64][S]
  unsigned short* cws = xb;                      // ctx reuses xb

  dim3 blk(256);
  dim3 gT(16, 16);
  transpose_w_bf16<<<gT, blk, 0, stream>>>(Wq, wt + 0 * KK);
  transpose_w_bf16<<<gT, blk, 0, stream>>>(Wk, wt + 1 * KK);
  transpose_w_bf16<<<gT, blk, 0, stream>>>(Wv, wt + 2 * KK);
  transpose_w_bf16<<<gT, blk, 0, stream>>>(Wo, wt + 3 * KK);

  dim3 gG(M / 128, N / 64);
  const int nCvt = (int)(MK / (256 * 8));
  cvt_f32_bf16<<<nCvt, blk, 0, stream>>>(query, xb);
  gemm_bf16<<<gG, blk, 0, stream>>>(xb, wt + 0 * KK, bq, qws, 1, M, N, K);
  cvt_f32_bf16<<<nCvt, blk, 0, stream>>>(key, xb);
  gemm_bf16<<<gG, blk, 0, stream>>>(xb, wt + 1 * KK, bk, kws, 1, M, N, K);
  cvt_f32_bf16<<<nCvt, blk, 0, stream>>>(value, xb);
  gemm_bf16<<<gG, blk, 0, stream>>>(xb, wt + 2 * KK, bv, vws, 2, M, N, K);

  flash_attn<<<dim3(64 * 16), blk, 0, stream>>>(qws, kws, vws, cws);

  gemm_bf16<<<gG, blk, 0, stream>>>(cws, wt + 3 * KK, bo, d_out, 0, M, N, K);
}
